// GAT_20761871909628
// MI455X (gfx1250) — matrix operand order repaired and checked by reading, not compiled, not run on hardware
//
#include <hip/hip_runtime.h>

#define NN    50000
#define NPAD  50048    // NN rounded up to 128 (GEMM1 M-tile)
#define E0    800000
#define ET    850000   // with self loops
#define FIN   256
#define HEADS 8
#define HID   64
#define HC    512      // HEADS*HID
#define NCLS  16

typedef __attribute__((ext_vector_type(16))) __bf16 v16bf;
typedef __attribute__((ext_vector_type(8)))  __bf16 v8bf;
typedef __attribute__((ext_vector_type(8)))  float  v8f;

__device__ __forceinline__ float atomicMaxF(float* addr, float val) {
  if (val >= 0.0f)
    return __int_as_float(atomicMax((int*)addr, __float_as_int(val)));
  else
    return __uint_as_float(atomicMin((unsigned int*)addr, __float_as_uint(val)));
}

__device__ __forceinline__ float lrelu(float x) { return x > 0.0f ? x : 0.2f * x; }

__device__ __forceinline__ void edge_sd(const int* __restrict__ ei, int e, int& s, int& d) {
  if (e < E0) { s = ei[e]; d = ei[E0 + e]; } else { s = e - E0; d = e - E0; }
}

__device__ __forceinline__ v16bf cat16(v8bf lo, v8bf hi) {
  return __builtin_shufflevector(lo, hi, 0,1,2,3,4,5,6,7,8,9,10,11,12,13,14,15);
}

// Async copy: each lane moves 16B global -> LDS. Tracked by ASYNCcnt.
__device__ __forceinline__ void async_cp16(const __bf16* lds_ptr, const __bf16* gptr) {
  unsigned lds = (unsigned)(unsigned long long)(const void*)lds_ptr;  // LDS byte offset
  unsigned long long ga = (unsigned long long)gptr;
  asm volatile("global_load_async_to_lds_b128 %0, %1, off"
               :: "v"(lds), "v"(ga) : "memory");
}

// ---------------------------------------------------------------- fill
__global__ void k_fill(float* __restrict__ p, int n, float v) {
  int i = blockIdx.x * blockDim.x + threadIdx.x;
  if (i < n) p[i] = v;
}

// ---------------------------------------------------------------- f32 -> bf16 pre-conversions
__global__ void k_cvtx(const float* __restrict__ X, __bf16* __restrict__ Xbf) {
  int i = blockIdx.x * blockDim.x + threadIdx.x;
  if (i >= NPAD * FIN) return;
  Xbf[i] = (i < NN * FIN) ? (__bf16)X[i] : (__bf16)0.0f;   // zero the M padding rows
}

__global__ void k_cvtw1(const float* __restrict__ W, __bf16* __restrict__ Wt) {
  int i = blockIdx.x * blockDim.x + threadIdx.x;
  if (i >= FIN * HC) return;
  const int k = i >> 9, n = i & (HC - 1);
  Wt[(size_t)n * FIN + k] = (__bf16)W[i];                  // W1^T : [n][k]
}

// ---------------------------------------------------------------- GEMM1: h1 = x @ W1 (bf16 WMMA, async-LDS double buffered)
// M=50048(pad), K=256, N=512. Block 256 thr = 8 waves; tile 128x128, K-step 32.
__global__ __launch_bounds__(256)
void k_gemm1(const __bf16* __restrict__ Xbf, const __bf16* __restrict__ W1t,
             __bf16* __restrict__ Hbf) {
  __shared__ __align__(16) __bf16 Al[2][128][40];  // A tile 128x32, stride 40 halves = 80B
  __shared__ __align__(16) __bf16 Bt[2][128][40];  // B tile (already n-major) 128x32
  const int t    = threadIdx.x;
  const int lane = t & 31;
  const int wv   = t >> 5;
  const int wm   = wv & 3;        // 4 waves along M (32 rows each)
  const int wn   = wv >> 2;       // 2 waves along N (64 cols each)
  const int m0   = blockIdx.x * 128;
  const int n0   = blockIdx.y * 128;
  const int l15  = lane & 15;
  const int kg   = lane >> 4;

  v8f acc[2][4];
  #pragma unroll
  for (int i = 0; i < 2; ++i)
    #pragma unroll
    for (int j = 0; j < 4; ++j)
      #pragma unroll
      for (int r = 0; r < 8; ++r) acc[i][j][r] = 0.0f;

  // issue one 128x32 A tile + 128x32 B tile as async b128 copies (4 per thread / per wave)
  auto issue = [&](int ks, int buf) {
    const int k0 = ks * 32;
    #pragma unroll
    for (int p = 0; p < 2; ++p) {
      const int u = t + p * 256;          // 512 16B-units per tile
      const int r = u >> 2, ch = u & 3;   // row, 16B chunk (8 halves)
      async_cp16(&Al[buf][r][ch * 8], Xbf + (size_t)(m0 + r) * FIN + k0 + ch * 8);
      async_cp16(&Bt[buf][r][ch * 8], W1t + (size_t)(n0 + r) * FIN + k0 + ch * 8);
    }
  };

  issue(0, 0);
  for (int ks = 0; ks < FIN / 32; ++ks) {
    const int buf = ks & 1;
    if (ks < FIN / 32 - 1) {
      issue(ks + 1, buf ^ 1);                               // prefetch next tile
      asm volatile("s_wait_asynccnt 4" ::: "memory");       // older tile's 4 ops done
    } else {
      asm volatile("s_wait_asynccnt 0" ::: "memory");
    }
    __syncthreads();

    v16bf afr[2], bfr[4];
    #pragma unroll
    for (int i = 0; i < 2; ++i) {
      const int row = wm * 32 + i * 16 + l15;
      v8bf lo = *(const v8bf*)&Al[buf][row][kg * 8];        // K = kg*8 + 0..7
      v8bf hi = *(const v8bf*)&Al[buf][row][16 + kg * 8];   // K = 16 + kg*8 + 0..7
      afr[i] = cat16(lo, hi);
    }
    #pragma unroll
    for (int j = 0; j < 4; ++j) {
      const int col = wn * 64 + j * 16 + l15;
      v8bf lo = *(const v8bf*)&Bt[buf][col][kg * 8];        // K = kg*8 + 0..7
      v8bf hi = *(const v8bf*)&Bt[buf][col][16 + kg * 8];   // K = 16 + kg*8 + 0..7
      bfr[j] = cat16(lo, hi);
    }
    #pragma unroll
    for (int i = 0; i < 2; ++i)
      #pragma unroll
      for (int j = 0; j < 4; ++j)
        acc[i][j] = __builtin_amdgcn_wmma_f32_16x16x32_bf16(
            false, afr[i], false, bfr[j], (short)0, acc[i][j], false, false);
    __syncthreads();
  }

  #pragma unroll
  for (int i = 0; i < 2; ++i)
    #pragma unroll
    for (int j = 0; j < 4; ++j)
      #pragma unroll
      for (int r = 0; r < 8; ++r) {
        const int row = m0 + wm * 32 + i * 16 + kg * 8 + r;
        const int col = n0 + wn * 64 + j * 16 + l15;
        if (row < NN) Hbf[(size_t)row * HC + col] = (__bf16)acc[i][j][r];
      }
}

// ---------------------------------------------------------------- attention coefficients layer 1
__global__ void k_att1(const __bf16* __restrict__ H, const float* __restrict__ as,
                       const float* __restrict__ ad, float* __restrict__ es, float* __restrict__ ed) {
  int i = blockIdx.x * blockDim.x + threadIdx.x;
  if (i >= NN * HEADS) return;
  const int n = i >> 3, h = i & 7;
  const __bf16* hp = H + (size_t)n * HC + h * HID;
  const float*  ap = as + h * HID;
  const float*  bp = ad + h * HID;
  float s1 = 0.f, s2 = 0.f;
  #pragma unroll 8
  for (int j = 0; j < HID; ++j) { float v = (float)hp[j]; s1 += v * ap[j]; s2 += v * bp[j]; }
  es[i] = s1; ed[i] = s2;
}

// ---------------------------------------------------------------- edge passes layer 1
__global__ void k_max1(const int* __restrict__ ei, const float* __restrict__ es,
                       const float* __restrict__ ed, float* __restrict__ m1) {
  int i = blockIdx.x * blockDim.x + threadIdx.x;
  if (i >= ET * HEADS) return;
  const int e = i >> 3, h = i & 7;
  int s, d; edge_sd(ei, e, s, d);
  float sc = lrelu(es[s * HEADS + h] + ed[d * HEADS + h]);
  atomicMaxF(&m1[d * HEADS + h], sc);
}

__global__ void k_sum1(const int* __restrict__ ei, const float* __restrict__ es,
                       const float* __restrict__ ed, const float* __restrict__ m1,
                       float* __restrict__ dn1) {
  int i = blockIdx.x * blockDim.x + threadIdx.x;
  if (i >= ET * HEADS) return;
  const int e = i >> 3, h = i & 7;
  int s, d; edge_sd(ei, e, s, d);
  float sc = lrelu(es[s * HEADS + h] + ed[d * HEADS + h]);
  atomicAdd(&dn1[d * HEADS + h], __expf(sc - m1[d * HEADS + h]));
}

__global__ __launch_bounds__(256)
void k_agg1(const int* __restrict__ ei, const __bf16* __restrict__ H,
            const float* __restrict__ es, const float* __restrict__ ed,
            const float* __restrict__ m1, const float* __restrict__ dn1,
            float* __restrict__ out1) {
  const int gid = blockIdx.x * 256 + threadIdx.x;
  const int e = gid >> 6;     // 64 threads / edge (one per channel)
  const int c = gid & 63;
  if (e >= ET) return;
  int s, d; edge_sd(ei, e, s, d);
  const __bf16* hs = H + (size_t)s * HC;
  float* od = out1 + (size_t)d * HC;
  #pragma unroll
  for (int h = 0; h < HEADS; ++h) {
    float sc    = lrelu(es[s * HEADS + h] + ed[d * HEADS + h]);
    float alpha = __expf(sc - m1[d * HEADS + h]) / (dn1[d * HEADS + h] + 1e-16f);
    atomicAdd(&od[h * HID + c], alpha * (float)hs[h * HID + c]);
  }
}

// ---------------------------------------------------------------- ELU (+b1) -> bf16 features for layer 2
__global__ void k_elu(const float* __restrict__ out1, const float* __restrict__ b1,
                      __bf16* __restrict__ h2) {
  int i = blockIdx.x * blockDim.x + threadIdx.x;
  if (i >= NN * HC) return;
  float v = out1[i] + b1[i & (HC - 1)];
  float r = v > 0.0f ? v : (__expf(v) - 1.0f);
  h2[i] = (__bf16)r;
}

// ---------------------------------------------------------------- GEMM2: g2 = h2 @ W2 (bf16 WMMA)
// M=50000, K=512, N=16. 8 waves * 16 rows = 128 rows / block. W2^T cached in LDS.
__global__ __launch_bounds__(256)
void k_gemm2(const __bf16* __restrict__ H2, const float* __restrict__ W2, float* __restrict__ G2) {
  __shared__ __bf16 Bt[NCLS][520];  // [n][k], stride 520 halves = 1040B (16B aligned, bank-spread)
  const int t = threadIdx.x;
  for (int idx = t; idx < HC * NCLS; idx += 256) {
    const int n = idx & 15, k = idx >> 4;
    Bt[n][k] = (__bf16)W2[idx];
  }
  __syncthreads();

  const int lane = t & 31, wv = t >> 5;
  const int l15 = lane & 15, kg = lane >> 4;
  const int rbase = blockIdx.x * 128 + wv * 16;
  int rowA = rbase + l15;
  if (rowA >= NN) rowA = NN - 1;
  const __bf16* arow = H2 + (size_t)rowA * HC;

  v8f acc;
  #pragma unroll
  for (int r = 0; r < 8; ++r) acc[r] = 0.0f;

  for (int ks = 0; ks < HC / 32; ++ks) {
    const int k0 = ks * 32;
    v8bf alo = *(const v8bf*)(arow + k0 + kg * 8);
    v8bf ahi = *(const v8bf*)(arow + k0 + 16 + kg * 8);
    v16bf a = cat16(alo, ahi);
    v8bf blo = *(const v8bf*)&Bt[l15][k0 + kg * 8];
    v8bf bhi = *(const v8bf*)&Bt[l15][k0 + 16 + kg * 8];
    v16bf b = cat16(blo, bhi);
    acc = __builtin_amdgcn_wmma_f32_16x16x32_bf16(false, a, false, b, (short)0, acc, false, false);
  }
  #pragma unroll
  for (int r = 0; r < 8; ++r) {
    const int row = rbase + kg * 8 + r;
    if (row < NN) G2[(size_t)row * NCLS + l15] = acc[r];
  }
}

// ---------------------------------------------------------------- attention coefficients layer 2 (1 head, 16 ch)
__global__ void k_att2(const float* __restrict__ G2, const float* __restrict__ as2,
                       const float* __restrict__ ad2, float* __restrict__ e2s, float* __restrict__ e2d) {
  int n = blockIdx.x * blockDim.x + threadIdx.x;
  if (n >= NN) return;
  const float* g = G2 + (size_t)n * NCLS;
  float s1 = 0.f, s2 = 0.f;
  #pragma unroll
  for (int c = 0; c < NCLS; ++c) { s1 += g[c] * as2[c]; s2 += g[c] * ad2[c]; }
  e2s[n] = s1; e2d[n] = s2;
}

__global__ void k_max2(const int* __restrict__ ei, const float* __restrict__ e2s,
                       const float* __restrict__ e2d, float* __restrict__ m2) {
  int e = blockIdx.x * blockDim.x + threadIdx.x;
  if (e >= ET) return;
  int s, d; edge_sd(ei, e, s, d);
  atomicMaxF(&m2[d], lrelu(e2s[s] + e2d[d]));
}

__global__ void k_sum2(const int* __restrict__ ei, const float* __restrict__ e2s,
                       const float* __restrict__ e2d, const float* __restrict__ m2,
                       float* __restrict__ dn2) {
  int e = blockIdx.x * blockDim.x + threadIdx.x;
  if (e >= ET) return;
  int s, d; edge_sd(ei, e, s, d);
  float sc = lrelu(e2s[s] + e2d[d]);
  atomicAdd(&dn2[d], __expf(sc - m2[d]));
}

__global__ __launch_bounds__(256)
void k_agg2(const int* __restrict__ ei, const float* __restrict__ G2,
            const float* __restrict__ e2s, const float* __restrict__ e2d,
            const float* __restrict__ m2, const float* __restrict__ dn2,
            float* __restrict__ out2) {
  const int gid = blockIdx.x * 256 + threadIdx.x;
  const int e = gid >> 4;     // 16 threads / edge
  const int c = gid & 15;
  if (e >= ET) return;
  int s, d; edge_sd(ei, e, s, d);
  float sc    = lrelu(e2s[s] + e2d[d]);
  float alpha = __expf(sc - m2[d]) / (dn2[d] + 1e-16f);
  atomicAdd(&out2[(size_t)d * NCLS + c], alpha * G2[(size_t)s * NCLS + c]);
}

// ---------------------------------------------------------------- +b2, log_softmax
__global__ void k_lsm(const float* __restrict__ out2, const float* __restrict__ b2,
                      float* __restrict__ O) {
  int n = blockIdx.x * blockDim.x + threadIdx.x;
  if (n >= NN) return;
  float l[NCLS];
  float mx = -3.4e38f;
  #pragma unroll
  for (int c = 0; c < NCLS; ++c) { l[c] = out2[(size_t)n * NCLS + c] + b2[c]; mx = fmaxf(mx, l[c]); }
  float sum = 0.f;
  #pragma unroll
  for (int c = 0; c < NCLS; ++c) sum += __expf(l[c] - mx);
  float ls = __logf(sum);
  #pragma unroll
  for (int c = 0; c < NCLS; ++c) O[(size_t)n * NCLS + c] = l[c] - mx - ls;
}

// ---------------------------------------------------------------- launch
extern "C" void kernel_launch(void* const* d_in, const int* in_sizes, int n_in,
                              void* d_out, int out_size, void* d_ws, size_t ws_size,
                              hipStream_t stream) {
  const float* x    = (const float*)d_in[0];
  const int*   ei   = (const int*)d_in[1];
  const float* W1   = (const float*)d_in[2];
  const float* as1  = (const float*)d_in[3];
  const float* ad1  = (const float*)d_in[4];
  const float* b1   = (const float*)d_in[5];
  const float* W2   = (const float*)d_in[6];
  const float* as2  = (const float*)d_in[7];
  const float* ad2  = (const float*)d_in[8];
  const float* b2   = (const float*)d_in[9];
  float* O = (float*)d_out;

  char* ws = (char*)d_ws;
  size_t off = 0;
  auto carve = [&](size_t bytes) { char* p = ws + off; off += (bytes + 255) & ~(size_t)255; return p; };
  __bf16* xbf  = (__bf16*)carve((size_t)NPAD * FIN * 2);
  __bf16* w1t  = (__bf16*)carve((size_t)HC * FIN * 2);
  __bf16* h1bf = (__bf16*)carve((size_t)NN * HC * 2);
  float*  es1  = (float*)carve((size_t)NN * HEADS * 4);
  float*  ed1  = (float*)carve((size_t)NN * HEADS * 4);
  float*  m1   = (float*)carve((size_t)NN * HEADS * 4);
  float*  dn1  = (float*)carve((size_t)NN * HEADS * 4);
  float*  out1 = (float*)carve((size_t)NN * HC * 4);
  __bf16* h2bf = (__bf16*)carve((size_t)NN * HC * 2);
  float*  g2   = (float*)carve((size_t)NN * NCLS * 4);
  float*  e2s  = (float*)carve((size_t)NN * 4);
  float*  e2d  = (float*)carve((size_t)NN * 4);
  float*  m2   = (float*)carve((size_t)NN * 4);
  float*  dn2  = (float*)carve((size_t)NN * 4);
  float*  out2 = (float*)carve((size_t)NN * NCLS * 4);
  (void)ws_size; (void)n_in; (void)in_sizes; (void)out_size;

  const float NEG_INF = -__builtin_huge_valf();
  auto blocks = [](long long n, int bs) { return (unsigned)((n + bs - 1) / bs); };

  // init accumulators / reductions
  k_fill<<<blocks((long long)NN * HEADS, 256), 256, 0, stream>>>(m1, NN * HEADS, NEG_INF);
  k_fill<<<blocks((long long)NN * HEADS, 256), 256, 0, stream>>>(dn1, NN * HEADS, 0.0f);
  k_fill<<<blocks((long long)NN * HC, 256), 256, 0, stream>>>(out1, NN * HC, 0.0f);
  k_fill<<<blocks(NN, 256), 256, 0, stream>>>(m2, NN, NEG_INF);
  k_fill<<<blocks(NN, 256), 256, 0, stream>>>(dn2, NN, 0.0f);
  k_fill<<<blocks((long long)NN * NCLS, 256), 256, 0, stream>>>(out2, NN * NCLS, 0.0f);

  // bf16 staging for async-LDS GEMM1
  k_cvtx<<<blocks((long long)NPAD * FIN, 256), 256, 0, stream>>>(x, xbf);
  k_cvtw1<<<blocks((long long)FIN * HC, 256), 256, 0, stream>>>(W1, w1t);

  // layer 1
  dim3 g1(blocks(NPAD, 128), HC / 128);
  k_gemm1<<<g1, 256, 0, stream>>>(xbf, w1t, h1bf);
  k_att1<<<blocks((long long)NN * HEADS, 256), 256, 0, stream>>>(h1bf, as1, ad1, es1, ed1);
  k_max1<<<blocks((long long)ET * HEADS, 256), 256, 0, stream>>>(ei, es1, ed1, m1);
  k_sum1<<<blocks((long long)ET * HEADS, 256), 256, 0, stream>>>(ei, es1, ed1, m1, dn1);
  k_agg1<<<blocks((long long)ET * 64, 256), 256, 0, stream>>>(ei, h1bf, es1, ed1, m1, dn1, out1);
  k_elu<<<blocks((long long)NN * HC, 256), 256, 0, stream>>>(out1, b1, h2bf);

  // layer 2
  k_gemm2<<<blocks(NN, 128), 256, 0, stream>>>(h2bf, W2, g2);
  k_att2<<<blocks(NN, 256), 256, 0, stream>>>(g2, as2, ad2, e2s, e2d);
  k_max2<<<blocks(ET, 256), 256, 0, stream>>>(ei, e2s, e2d, m2);
  k_sum2<<<blocks(ET, 256), 256, 0, stream>>>(ei, e2s, e2d, m2, dn2);
  k_agg2<<<blocks((long long)ET * NCLS, 256), 256, 0, stream>>>(ei, g2, e2s, e2d, m2, dn2, out2);
  k_lsm<<<blocks(NN, 256), 256, 0, stream>>>(out2, b2, O);
}
